// MultiHeadSelfAttention_51711406243842
// MI455X (gfx1250) — matrix operand order repaired and checked by reading, not compiled, not run on hardware
//
#include <hip/hip_runtime.h>
#include <hip/hip_bf16.h>

typedef __attribute__((ext_vector_type(16))) __bf16 v16bf;
typedef __attribute__((ext_vector_type(8)))  float  v8f;
typedef __attribute__((ext_vector_type(4)))  int    v4i;

union Frag {
  v16bf v;
  unsigned u[8];
  unsigned short h[16];
};

__device__ __forceinline__ unsigned short f2bf(float f) {
  unsigned u = __float_as_uint(f);
  u += 0x7FFFu + ((u >> 16) & 1u);   // round-to-nearest-even
  return (unsigned short)(u >> 16);
}
__device__ __forceinline__ unsigned short to_bf16(float x) { return f2bf(x); }
__device__ __forceinline__ unsigned short to_bf16(unsigned short x) { return x; }

__device__ __forceinline__ unsigned ld_u32(const unsigned short* p) {
  unsigned r; __builtin_memcpy(&r, p, 4); return r;
}

#if defined(__gfx1250__) && __has_builtin(__builtin_amdgcn_global_load_async_to_lds_b128)
#define USE_ASYNC_LDS 1
typedef __attribute__((address_space(1))) v4i* gv4p;  // pointer to global v4i
typedef __attribute__((address_space(3))) v4i* lv4p;  // pointer to LDS v4i
#else
#define USE_ASYNC_LDS 0
#endif

// ---------------------------------------------------------------------------
// GEMM: C[M,N] = A[M,K] @ W[K,N] + bias,  A is f32 or bf16(ushort), W/bias f32
// workgroup tile 64x128, 8 waves, wave tile 16x64, K step 32 (bf16 WMMA)
// ---------------------------------------------------------------------------
template <typename AT, bool OUT_BF16>
__global__ __launch_bounds__(256)
void gemm_bias_wmma(const AT* __restrict__ A, const float* __restrict__ W,
                    const float* __restrict__ bias, void* __restrict__ Cout,
                    int M, int N, int K, int lda, int ldw, int ldc) {
  __shared__ unsigned short sA[64 * 34];    // [row][k], pad stride 34
  __shared__ unsigned short sBt[128 * 34];  // [col][k] (W transposed), pad 34

  const int tid = threadIdx.x;
  const int m0 = blockIdx.x * 64;
  const int n0 = blockIdx.y * 128;
  const int wid = tid >> 5, lane = tid & 31, li = lane & 15, lh = lane >> 4;
  const int wm = wid & 3, wn = wid >> 2;

  const v8f vz = {0.f,0.f,0.f,0.f,0.f,0.f,0.f,0.f};
  v8f acc[4] = {vz, vz, vz, vz};

  for (int k0 = 0; k0 < K; k0 += 32) {
    {
      int row = tid >> 2;
      int kb  = (tid & 3) * 8;
      const AT* src = A + (size_t)(m0 + row) * lda + k0 + kb;
      #pragma unroll
      for (int i = 0; i < 8; ++i) sA[row * 34 + kb + i] = to_bf16(src[i]);
    }
    {
      int kk = tid >> 3;
      int nb = (tid & 7) * 16;
      const float* src = W + (size_t)(k0 + kk) * ldw + n0 + nb;
      #pragma unroll
      for (int i = 0; i < 16; ++i) sBt[(nb + i) * 34 + kk] = f2bf(src[i]);
    }
    __syncthreads();

    Frag af;
    const unsigned short* abase = &sA[(wm * 16 + li) * 34];
    #pragma unroll
    for (int p = 0; p < 8; ++p) {   // A layout: K = 2p + 16*lh
      int kA = (p << 1) + (lh << 4);
      af.u[p] = ld_u32(abase + kA);
    }
    #pragma unroll
    for (int j = 0; j < 4; ++j) {
      Frag bfr;
      const unsigned short* bbase = &sBt[(wn * 64 + j * 16 + li) * 34];
      #pragma unroll
      for (int p = 0; p < 8; ++p) { // B layout: K = 2p + 16*lh
        int kB = (p << 1) + (lh << 4);
        bfr.u[p] = ld_u32(bbase + kB);
      }
      acc[j] = __builtin_amdgcn_wmma_f32_16x16x32_bf16(
          false, af.v, false, bfr.v, (short)0, acc[j], false, false);
    }
    __syncthreads();
  }

  #pragma unroll
  for (int j = 0; j < 4; ++j) {
    int c = n0 + wn * 64 + j * 16 + li;
    float bb = bias[c];
    #pragma unroll
    for (int e = 0; e < 8; ++e) {   // C layout: row = e + 8*lh, col = li
      int r = m0 + wm * 16 + e + (lh << 3);
      float val = acc[j][e] + bb;
      if (OUT_BF16) ((unsigned short*)Cout)[(size_t)r * ldc + c] = f2bf(val);
      else          ((float*)Cout)[(size_t)r * ldc + c] = val;
    }
  }
}

// ---------------------------------------------------------------------------
// Flash attention: one block per (b, head, 64-row tile); 4 waves x 16 rows.
// K tile staged into LDS (async-to-LDS when available); V tile staged
// transposed into LDS so PV B-fragments are contiguous 32B LDS reads.
// ---------------------------------------------------------------------------
#define LTS 72  // LDS row stride in halves: 144B rows -> 16B-aligned fragments

__global__ __launch_bounds__(128)
void attn_flash_wmma(const unsigned short* __restrict__ qw,
                     const unsigned short* __restrict__ kw,
                     const unsigned short* __restrict__ vw,
                     const unsigned char* __restrict__ mask,
                     unsigned short* __restrict__ o, int S) {
  __shared__ unsigned short kS[64 * LTS];        // kS[n][d]
  __shared__ unsigned short vT[64 * LTS];        // vT[d][n]  (V transposed)
  __shared__ unsigned short pShm[4][16][LTS];    // per-wave P tile 16x64

  const int tid = threadIdx.x;
  const int wid = tid >> 5, lane = tid & 31, li = lane & 15, lh = lane >> 4;
  const int m0 = blockIdx.x * 64 + wid * 16;
  const int hd = blockIdx.y;
  const int b  = blockIdx.z;
  const int HD = 1024;  // H * 64
  const size_t rowBase = ((size_t)b * S) * HD + (size_t)hd * 64;

  // Q fragments (16 rows x 64 d) in registers, A layout
  Frag qf[2];
  {
    const unsigned short* qrow = qw + rowBase + (size_t)(m0 + li) * HD;
    #pragma unroll
    for (int kc = 0; kc < 2; ++kc)
      #pragma unroll
      for (int p = 0; p < 8; ++p) {
        int d = kc * 32 + (p << 1) + (lh << 4);
        qf[kc].u[p] = ld_u32(qrow + d);
      }
  }

  float mrow[8], lrow[8];
  #pragma unroll
  for (int e = 0; e < 8; ++e) { mrow[e] = -INFINITY; lrow[e] = 0.f; }
  const v8f vz = {0.f,0.f,0.f,0.f,0.f,0.f,0.f,0.f};
  v8f acc[4] = {vz, vz, vz, vz};
  const float scale = 0.125f;  // 1/sqrt(64)

  for (int n0 = 0; n0 < S; n0 += 64) {
    if (n0 + 64 < S) {
      __builtin_prefetch(kw + rowBase + (size_t)(n0 + 64 + lane) * HD, 0, 1);
      __builtin_prefetch(vw + rowBase + (size_t)(n0 + 64 + lane) * HD, 0, 1);
    }
    // ---- stage K tile (64 rows x 64 halves): 4 x 16B per thread ----
    {
      int r = tid >> 1;
      int c = (tid & 1) * 32;
      const unsigned short* g = kw + rowBase + (size_t)(n0 + r) * HD + c;
      unsigned short* l = &kS[r * LTS + c];
#if USE_ASYNC_LDS
      #pragma unroll
      for (int i = 0; i < 4; ++i)
        __builtin_amdgcn_global_load_async_to_lds_b128(
            (gv4p)(uintptr_t)(g + i * 8),
            (lv4p)(unsigned)(uintptr_t)(l + i * 8),
            0, 0);
#else
      #pragma unroll
      for (int i = 0; i < 4; ++i) __builtin_memcpy(l + i * 8, g + i * 8, 16);
#endif
    }
    // ---- stage V tile transposed: 2x(2n x 8d) subblocks per thread ----
    #pragma unroll
    for (int s2 = 0; s2 < 2; ++s2) {
      int sb = (tid << 1) + s2;          // 0..255
      int n  = (sb >> 3) << 1;           // 0,2,..,62
      int dc = (sb & 7) * 8;             // 0,8,..,56
      const unsigned short* g0 = vw + rowBase + (size_t)(n0 + n) * HD + dc;
      unsigned short a0[8], a1[8];
      __builtin_memcpy(a0, g0, 16);
      __builtin_memcpy(a1, g0 + HD, 16);
      #pragma unroll
      for (int i = 0; i < 8; ++i) {
        unsigned pk = (unsigned)a0[i] | ((unsigned)a1[i] << 16);
        __builtin_memcpy(&vT[(dc + i) * LTS + n], &pk, 4);
      }
    }
#if USE_ASYNC_LDS
#if __has_builtin(__builtin_amdgcn_s_wait_asynccnt)
    __builtin_amdgcn_s_wait_asynccnt(0);
#else
    asm volatile("s_wait_asynccnt 0" ::: "memory");
#endif
#endif
    __syncthreads();

    // ---- S = Q @ K^T (B-fragments from kS: contiguous 32B LDS reads) ----
    v8f s[4];
    #pragma unroll
    for (int j = 0; j < 4; ++j) {
      v8f c = vz;
      const unsigned short* kbase = &kS[(j * 16 + li) * LTS + (lh << 4)];
      #pragma unroll
      for (int kc = 0; kc < 2; ++kc) {
        Frag kbf;
        #pragma unroll
        for (int p = 0; p < 8; ++p) kbf.u[p] = ld_u32(kbase + kc * 32 + (p << 1));
        c = __builtin_amdgcn_wmma_f32_16x16x32_bf16(
            false, qf[kc].v, false, kbf.v, (short)0, c, false, false);
      }
      unsigned char mv = mask[(size_t)b * S + n0 + j * 16 + li];
      #pragma unroll
      for (int e = 0; e < 8; ++e) c[e] = mv ? c[e] * scale : -INFINITY;
      s[j] = c;
    }
    // ---- online softmax (row = e + 8*lh, cols over 16 lanes x 4 subtiles)
    #pragma unroll
    for (int e = 0; e < 8; ++e) {
      float mx = fmaxf(fmaxf(s[0][e], s[1][e]), fmaxf(s[2][e], s[3][e]));
      #pragma unroll
      for (int off = 1; off < 16; off <<= 1) mx = fmaxf(mx, __shfl_xor(mx, off, 32));
      float mnew = fmaxf(mrow[e], mx);
      float p0 = __expf(s[0][e] - mnew);
      float p1 = __expf(s[1][e] - mnew);
      float p2 = __expf(s[2][e] - mnew);
      float p3 = __expf(s[3][e] - mnew);
      float ls = (p0 + p1) + (p2 + p3);
      #pragma unroll
      for (int off = 1; off < 16; off <<= 1) ls += __shfl_xor(ls, off, 32);
      float alpha = __expf(mrow[e] - mnew);
      lrow[e] = lrow[e] * alpha + ls;
      mrow[e] = mnew;
      #pragma unroll
      for (int j = 0; j < 4; ++j) acc[j][e] *= alpha;
      int r = e + (lh << 3);
      pShm[wid][r][ 0 + li] = f2bf(p0);
      pShm[wid][r][16 + li] = f2bf(p1);
      pShm[wid][r][32 + li] = f2bf(p2);
      pShm[wid][r][48 + li] = f2bf(p3);
    }
    // DS ops are in-order per wave; wait so cross-lane LDS reads see P
    asm volatile("s_wait_dscnt 0" ::: "memory");

    // ---- reload P as A fragments ----
    Frag pa[2];
    #pragma unroll
    for (int kc = 0; kc < 2; ++kc)
      #pragma unroll
      for (int p = 0; p < 8; ++p) {
        int k = kc * 32 + (p << 1) + (lh << 4);
        pa[kc].u[p] = ld_u32(&pShm[wid][li][k]);
      }
    // ---- O += P @ V (B-fragments from vT: contiguous 32B LDS reads) ----
    #pragma unroll
    for (int jo = 0; jo < 4; ++jo) {
      const unsigned short* vbase = &vT[(jo * 16 + li) * LTS + (lh << 4)];
      #pragma unroll
      for (int kc = 0; kc < 2; ++kc) {
        Frag vb;
        #pragma unroll
        for (int p = 0; p < 8; ++p) vb.u[p] = ld_u32(vbase + kc * 32 + (p << 1));
        acc[jo] = __builtin_amdgcn_wmma_f32_16x16x32_bf16(
            false, pa[kc].v, false, vb.v, (short)0, acc[jo], false, false);
      }
    }
    __syncthreads();  // before next tile overwrites kS / vT
  }

  #pragma unroll
  for (int jo = 0; jo < 4; ++jo) {
    int d = jo * 16 + li;
    #pragma unroll
    for (int e = 0; e < 8; ++e) {
      int r = m0 + e + (lh << 3);
      float inv = 1.0f / lrow[e];
      o[rowBase + (size_t)r * HD + d] = f2bf(acc[jo][e] * inv);
    }
  }
}

// ---------------------------------------------------------------------------
extern "C" void kernel_launch(void* const* d_in, const int* in_sizes, int n_in,
                              void* d_out, int out_size, void* d_ws, size_t ws_size,
                              hipStream_t stream) {
  (void)in_sizes; (void)n_in; (void)out_size; (void)ws_size;
  constexpr int Bb = 4, S = 2048, Dm = 1024, H = 16;
  constexpr int M = Bb * S;  // 8192

  const float*         q_in     = (const float*)d_in[0];
  const float*         k_in     = (const float*)d_in[1];
  const float*         v_in     = (const float*)d_in[2];
  const unsigned char* key_mask = (const unsigned char*)d_in[3];
  const float* Wq = (const float*)d_in[4];
  const float* bq = (const float*)d_in[5];
  const float* Wk = (const float*)d_in[6];
  const float* bk = (const float*)d_in[7];
  const float* Wv = (const float*)d_in[8];
  const float* bv = (const float*)d_in[9];
  const float* Wo = (const float*)d_in[10];
  const float* bo = (const float*)d_in[11];

  // workspace: qw | kw | vw | o  (bf16, 16 MB each)
  unsigned short* qw = (unsigned short*)d_ws;
  unsigned short* kw = qw + (size_t)M * Dm;
  unsigned short* vw = kw + (size_t)M * Dm;
  unsigned short* ow = vw + (size_t)M * Dm;

  dim3 gg(M / 64, Dm / 128);
  gemm_bias_wmma<float, true><<<gg, 256, 0, stream>>>(q_in, Wq, bq, qw, M, Dm, Dm, Dm, Dm, Dm);
  gemm_bias_wmma<float, true><<<gg, 256, 0, stream>>>(k_in, Wk, bk, kw, M, Dm, Dm, Dm, Dm, Dm);
  gemm_bias_wmma<float, true><<<gg, 256, 0, stream>>>(v_in, Wv, bv, vw, M, Dm, Dm, Dm, Dm, Dm);

  attn_flash_wmma<<<dim3(S / 64, H, Bb), 128, 0, stream>>>(qw, kw, vw, key_mask, ow, S);

  gemm_bias_wmma<unsigned short, false><<<gg, 256, 0, stream>>>(ow, Wo, bo, d_out, M, Dm, Dm, Dm, Dm, Dm);
}
